// PolicyNetwork_25220047962428
// MI455X (gfx1250) — matrix operand order repaired and checked by reading, not compiled, not run on hardware
//
#include <hip/hip_runtime.h>
#include <hip/hip_bf16.h>

typedef __attribute__((ext_vector_type(16))) _Float16 v16h;
typedef __attribute__((ext_vector_type(8)))  _Float16 v8h;
typedef __attribute__((ext_vector_type(4)))  _Float16 v4h;
typedef __attribute__((ext_vector_type(8)))  float    v8f;

namespace {
constexpr int kE = 9, kB = 16384, kIn = 512, kH1 = 512, kH2 = 256, kH3 = 128, kA = 16;
constexpr int kS = 528;                 // LDS activation row stride (halves), 16B-aligned rows
constexpr int kThreads = 256, kWaves = 8;
constexpr int kM = 64;                  // samples per block (4 row-tiles of 16)

// half-element offsets of each layer's swizzled f16 weights inside the Wt region
constexpr long kOff1 = 0;
constexpr long kOff2 = kOff1 + (long)kE * kH1 * kIn;   // 9*512*512
constexpr long kOff3 = kOff2 + (long)kE * kH2 * kH1;   // 9*256*512
constexpr long kOff4 = kOff3 + (long)kE * kH3 * kH2;   // 9*128*256

// int-indexed workspace layout (in ints, at base of d_ws)
constexpr int WS_CNT = 0;    // 9 counters
constexpr int WS_OFF = 16;   // 10 offsets
constexpr int WS_CUR = 32;   // 9 cursors
constexpr int WS_IDX = 48;   // 16384 routed sample ids
constexpr size_t WS_WT_BYTES = 1u << 17; // Wt region starts at +128 KB
}

__global__ void k_route_init(int* wsI) {
  if (threadIdx.x < kE) wsI[WS_CNT + threadIdx.x] = 0;
}

__global__ void k_count(const int* __restrict__ pos, int* __restrict__ wsI) {
  for (int i = blockIdx.x * blockDim.x + threadIdx.x; i < kB; i += gridDim.x * blockDim.x) {
    int p = pos[i]; p = p < 0 ? 0 : (p > 8 ? 8 : p);
    atomicAdd(&wsI[WS_CNT + p], 1);
  }
}

__global__ void k_scan(int* wsI) {
  if (blockIdx.x == 0 && threadIdx.x == 0) {
    int acc = 0;
    for (int e = 0; e < kE; ++e) {
      wsI[WS_OFF + e] = acc;
      wsI[WS_CUR + e] = acc;
      acc += wsI[WS_CNT + e];
    }
    wsI[WS_OFF + kE] = acc;
  }
}

__global__ void k_scatter(const int* __restrict__ pos, int* __restrict__ wsI) {
  for (int i = blockIdx.x * blockDim.x + threadIdx.x; i < kB; i += gridDim.x * blockDim.x) {
    int p = pos[i]; p = p < 0 ? 0 : (p > 8 ? 8 : p);
    int slot = atomicAdd(&wsI[WS_CUR + p], 1);
    wsI[WS_IDX + slot] = i;
  }
}

// Repack W[e][n][k] (f32, row-major [E,N,K]) into per-lane WMMA B-operand tiles (f16).
// Tile (e,nt,kt) = 32 lanes x 16 halves contiguous (1 KB). Within a k-tile of 32:
// lanes 0-15 hold K=kt*32+{0..7,16..23}, lanes 16-31 hold K=kt*32+{8..15,24..31}; lane%16 = N column.
__global__ void k_swizzle(const float* __restrict__ W, _Float16* __restrict__ Wt,
                          int N, int K, long loff) {
  const long total = (long)kE * N * K;
  const int NT = N >> 4, KT = K >> 5;
  for (long t = (long)blockIdx.x * blockDim.x + threadIdx.x; t < total;
       t += (long)gridDim.x * blockDim.x) {
    int k = (int)(t % K); long r = t / K; int n = (int)(r % N); int e = (int)(r / N);
    int nt = n >> 4, nn = n & 15, kt = k >> 5, kk = k & 31;
    int lane = nn + ((kk & 8) ? 16 : 0);
    int h = (kk & 7) | ((kk & 16) >> 1);
    long dst = loff + ((long)((e * NT + nt) * KT + kt) << 9) + lane * 16 + h;
    Wt[dst] = (_Float16)W[t];
  }
}

#define LOAD_A(dst, off)                                      \
  do {                                                        \
    (dst).h[0] = *(const v8h*)(ap + (off));                   \
    (dst).h[1] = *(const v8h*)(ap + (off) + 16);              \
  } while (0)

// 64-row tile: each wave owns an n-tile, reuses one B fragment across 4 row-tiles.
__device__ __forceinline__ void run_layer64(const _Float16* __restrict__ actIn,
                                            _Float16* __restrict__ actOut,
                                            const _Float16* __restrict__ wbase,
                                            const float* __restrict__ bias,
                                            int K, int N, int wave, int lane) {
  const int KT = K >> 5;
  const int ln = lane & 15, lh = (lane >> 4) << 3;
  for (int nt = wave; nt < (N >> 4); nt += kWaves) {
    v8f c0 = {}, c1 = {}, c2 = {}, c3 = {};
    const _Float16* bp = wbase + ((long)nt * KT << 9) + lane * 16;
    const _Float16* ap = actIn + ln * kS + lh;
#pragma unroll 4
    for (int kt = 0; kt < KT; ++kt) {
      v16h b = *(const v16h*)bp;          // one B fragment, 4x reuse below
      union { v16h v; v8h h[2]; } a0, a1, a2, a3;
      LOAD_A(a0, 0);
      LOAD_A(a1, 16 * kS);
      LOAD_A(a2, 32 * kS);
      LOAD_A(a3, 48 * kS);
      c0 = __builtin_amdgcn_wmma_f32_16x16x32_f16(false, a0.v, false, b, (short)0, c0, false, false);
      c1 = __builtin_amdgcn_wmma_f32_16x16x32_f16(false, a1.v, false, b, (short)0, c1, false, false);
      c2 = __builtin_amdgcn_wmma_f32_16x16x32_f16(false, a2.v, false, b, (short)0, c2, false, false);
      c3 = __builtin_amdgcn_wmma_f32_16x16x32_f16(false, a3.v, false, b, (short)0, c3, false, false);
      ap += 32;
      bp += 512;
    }
    const int col = (nt << 4) + ln;
    const float bv = bias[col];
#pragma unroll
    for (int r = 0; r < 8; ++r) {
      const int row = lh + r;
      actOut[(row)      * kS + col] = (_Float16)fmaxf(c0[r] + bv, 0.0f);
      actOut[(row + 16) * kS + col] = (_Float16)fmaxf(c1[r] + bv, 0.0f);
      actOut[(row + 32) * kS + col] = (_Float16)fmaxf(c2[r] + bv, 0.0f);
      actOut[(row + 48) * kS + col] = (_Float16)fmaxf(c3[r] + bv, 0.0f);
    }
  }
}

__global__ __launch_bounds__(kThreads) void k_moe(
    const float* __restrict__ x, const int* __restrict__ wsI,
    const _Float16* __restrict__ Wt,
    const float* __restrict__ b1, const float* __restrict__ b2,
    const float* __restrict__ b3, const float* __restrict__ b4,
    float* __restrict__ out) {
  __shared__ __align__(16) _Float16 actA[kM * kS];
  __shared__ __align__(16) _Float16 actB[kM * kS];
  __shared__ float logits[kM * 17];
  __shared__ int sid[kM];

  // Uniform device-side block -> (expert, tile) mapping.
  int e = -1, t = 0, acc = 0;
  for (int i = 0; i < kE; ++i) {
    int tiles = (wsI[WS_CNT + i] + kM - 1) >> 6;
    if (e < 0 && (int)blockIdx.x < acc + tiles) { e = i; t = (int)blockIdx.x - acc; }
    acc += tiles;
  }
  if (e < 0) return;

  const int cntE = wsI[WS_CNT + e];
  const int base = wsI[WS_OFF + e] + (t << 6);
  const int m = min(kM, cntE - (t << 6));
  const int tid = threadIdx.x;

  if (tid < kM) sid[tid] = (tid < m) ? wsI[WS_IDX + base + tid] : -1;
  __syncthreads();

  // Gather 64 routed rows of x (float4 loads), f32 -> f16, into LDS.
  for (int q = tid; q < kM * (kIn / 4); q += kThreads) {
    const int r = q >> 7, c4 = (q & 127) << 2;
    const int s = sid[r];
    v4h hv = {(_Float16)0.0f, (_Float16)0.0f, (_Float16)0.0f, (_Float16)0.0f};
    if (s >= 0) {
      const float4 v = *(const float4*)(x + (long)s * kIn + c4);
      hv = (v4h){(_Float16)v.x, (_Float16)v.y, (_Float16)v.z, (_Float16)v.w};
    }
    *(v4h*)(actA + r * kS + c4) = hv;
  }
  __syncthreads();

  const int wave = tid >> 5, lane = tid & 31;

  run_layer64(actA, actB, Wt + kOff1 + ((long)e * (kH1 >> 4) * (kIn >> 5) << 9),
              b1 + e * kH1, kIn, kH1, wave, lane);
  __syncthreads();
  run_layer64(actB, actA, Wt + kOff2 + ((long)e * (kH2 >> 4) * (kH1 >> 5) << 9),
              b2 + e * kH2, kH1, kH2, wave, lane);
  __syncthreads();
  run_layer64(actA, actB, Wt + kOff3 + ((long)e * (kH3 >> 4) * (kH2 >> 5) << 9),
              b3 + e * kH3, kH2, kH3, wave, lane);
  __syncthreads();

  // Layer 4: N=16 -> one n-tile; waves 0..3 each take one 16-row tile. No ReLU.
  if (wave < 4) {
    const int ln = lane & 15, lh = (lane >> 4) << 3;
    v8f c = {};
    const int KT = kH3 >> 5;
    const _Float16* bp = Wt + kOff4 + ((long)e * KT << 9) + lane * 16;
    const _Float16* ap = actB + (wave * 16 + ln) * kS + lh;
#pragma unroll
    for (int kt = 0; kt < KT; ++kt) {
      union { v16h v; v8h h[2]; } a;
      a.h[0] = *(const v8h*)(ap);
      a.h[1] = *(const v8h*)(ap + 16);
      v16h b = *(const v16h*)(bp);
      c = __builtin_amdgcn_wmma_f32_16x16x32_f16(false, a.v, false, b, (short)0, c, false, false);
      ap += 32;
      bp += 512;
    }
    const float bv = b4[e * kA + ln];
#pragma unroll
    for (int r = 0; r < 8; ++r) logits[(wave * 16 + lh + r) * 17 + ln] = c[r] + bv;
  }
  __syncthreads();

  // Softmax over 16 actions, one row per thread, scatter to output.
  if (tid < kM && tid < m) {
    float l[16], mx = -3.4e38f;
#pragma unroll
    for (int j = 0; j < 16; ++j) { l[j] = logits[tid * 17 + j]; mx = fmaxf(mx, l[j]); }
    float s = 0.0f;
#pragma unroll
    for (int j = 0; j < 16; ++j) { l[j] = __expf(l[j] - mx); s += l[j]; }
    const float inv = 1.0f / s;
    const long o = (long)sid[tid] * kA;
#pragma unroll
    for (int j = 0; j < 16; ++j) out[o + j] = l[j] * inv;
  }
}

extern "C" void kernel_launch(void* const* d_in, const int* in_sizes, int n_in,
                              void* d_out, int out_size, void* d_ws, size_t ws_size,
                              hipStream_t stream) {
  (void)in_sizes; (void)n_in; (void)out_size; (void)ws_size;
  const float* x   = (const float*)d_in[0];
  const int*   pos = (const int*)d_in[1];
  const float* W1  = (const float*)d_in[2];
  const float* bb1 = (const float*)d_in[3];
  const float* W2  = (const float*)d_in[4];
  const float* bb2 = (const float*)d_in[5];
  const float* W3  = (const float*)d_in[6];
  const float* bb3 = (const float*)d_in[7];
  const float* W4  = (const float*)d_in[8];
  const float* bb4 = (const float*)d_in[9];
  float* out = (float*)d_out;

  int* wsI = (int*)d_ws;
  _Float16* Wt = (_Float16*)((char*)d_ws + WS_WT_BYTES);

  // Routing (re-done every call: no cross-call state).
  k_route_init<<<1, 32, 0, stream>>>(wsI);
  k_count<<<64, kThreads, 0, stream>>>(pos, wsI);
  k_scan<<<1, 1, 0, stream>>>(wsI);
  k_scatter<<<64, kThreads, 0, stream>>>(pos, wsI);

  // Weight repack f32 -> swizzled f16 B-operand tiles.
  k_swizzle<<<512, kThreads, 0, stream>>>(W1, Wt, kH1, kIn, kOff1);
  k_swizzle<<<256, kThreads, 0, stream>>>(W2, Wt, kH2, kH1, kOff2);
  k_swizzle<<<64,  kThreads, 0, stream>>>(W3, Wt, kH3, kH2, kOff3);
  k_swizzle<<<8,   kThreads, 0, stream>>>(W4, Wt, kA,  kH3, kOff4);

  // Worst case sum_e ceil(cnt_e/64) <= 256 + 9; extra blocks exit early.
  k_moe<<<265, kThreads, 0, stream>>>(x, wsI, Wt, bb1, bb2, bb3, bb4, out);
}
